// Comm_27101243638533
// MI455X (gfx1250) — matrix operand order repaired and checked by reading, not compiled, not run on hardware
//
#include <hip/hip_runtime.h>
#include <math.h>

typedef __attribute__((ext_vector_type(16))) _Float16 v16h;
typedef __attribute__((ext_vector_type(8)))  _Float16 v8h;
typedef __attribute__((ext_vector_type(8)))  float    v8f;
typedef __attribute__((ext_vector_type(4)))  float    v4f;

#define AGENTS 8
#define VOCAB  32000
#define EMB    256
#define HID    512
#define NACT   6
#define FEAT   2048
#define LSTMIN 2311   // 2048 + 256 + 7

__device__ __forceinline__ float sigmoidf_(float x) { return 1.f / (1.f + expf(-x)); }

// ---------------- misc ----------------
__global__ void zero_kernel(float* p, int n) {
    int i = blockIdx.x * blockDim.x + threadIdx.x;
    if (i < n) p[i] = 0.f;
}

// ---------------- prev-message embedding: y[a] = m_prev[a] @ emb_w[a]  ([1,V]@[V,E]) ----------------
__global__ __launch_bounds__(256) void emb_kernel(const float* __restrict__ mprev,
                                                  const float* __restrict__ embw,
                                                  float* __restrict__ prev_emb) {
    int a = blockIdx.y;
    int t = threadIdx.x;
    const float* x = mprev + (long long)a * VOCAB;
    const float* W = embw + (long long)a * VOCAB * EMB;
    __shared__ float xs[256];
    float acc = 0.f;
    int v0 = blockIdx.x * (VOCAB / 8);          // 4000 rows per chunk
    int v1 = v0 + (VOCAB / 8);
    for (int vb = v0; vb < v1; vb += 256) {
        int nv = v1 - vb; if (nv > 256) nv = 256;
        __syncthreads();
        if (t < nv) xs[t] = x[vb + t];
        __syncthreads();
        for (int v = 0; v < nv; ++v)
            acc += xs[v] * __builtin_nontemporal_load(&W[(long long)(vb + v) * EMB + t]);
    }
    atomicAdd(&prev_emb[a * EMB + t], acc);
}

// ---------------- conv weights fp32 -> f16, zero-padded to Kpad: w16[9][16][Kpad] ----------------
__global__ __launch_bounds__(256) void wcvt_kernel(const float* __restrict__ aw,
                                                   const float* __restrict__ ow,
                                                   _Float16* __restrict__ w16,
                                                   int K, int Kpad) {
    int t = blockIdx.x * 256 + threadIdx.x;     // over 9*16*Kpad (exact multiple of 256)
    int img = t / (16 * Kpad);
    int rem = t % (16 * Kpad);
    int m = rem / Kpad, k = rem % Kpad;
    float v = 0.f;
    if (k < K) v = (img < 8) ? aw[(long long)img * 16 * K + m * K + k] : ow[m * K + k];
    w16[t] = (_Float16)v;
}

// ---------------- im2col (transposed, f16, zero-padded): bt[9][N][Kpad] ----------------
// All bounds checks live here; the WMMA GEMM sees only dense aligned data.
__global__ __launch_bounds__(256) void im2col_kernel(
        const float* __restrict__ inA, long long inStride, const float* __restrict__ inO,
        _Float16* __restrict__ bt,
        int Cin, int Hin, int Win, int Wout, int K, int Kpad, int N) {
    int img = blockIdx.y;
    int t = blockIdx.x * 256 + threadIdx.x;     // over N*Kpad (exact multiple of 256)
    int n = t / Kpad, k = t % Kpad;
    const float* in = (img < 8) ? (inA + (long long)img * inStride) : inO;
    float v = 0.f;
    if (k < K) {
        int cin = k / 9, rem = k % 9;
        int kh = rem / 3, kw = rem % 3;
        int oy = n / Wout, ox = n % Wout;
        int iy = oy * 2 - 1 + kh, ix = ox * 2 - 1 + kw;
        if (iy >= 0 && iy < Hin && ix >= 0 && ix < Win)
            v = in[(long long)cin * Hin * Win + iy * Win + ix];
    }
    bt[(long long)img * N * Kpad + t] = (_Float16)v;
}

// ---------------- GEMM + bias + CReLU via v_wmma_f32_16x16x32_f16 ----------------
// Block = 256 (8 waves); each wave owns one 16x16 C tile (M=16 out-chans, N=16 pixels).
// A fragment: two b128 loads from w16 row; B fragment: two b128 loads from transposed im2col row.
__global__ __launch_bounds__(256) void gemm_crelu_wmma(
        const _Float16* __restrict__ w16,   // [9][16][Kpad]
        const _Float16* __restrict__ bt,    // [9][N][Kpad]
        const float* __restrict__ ab, const float* __restrict__ ob,
        float* __restrict__ out,            // [9][32][N]
        int N, int Kpad) {
    const int img  = blockIdx.y;
    const int wave = threadIdx.x >> 5;
    const int lane = threadIdx.x & 31;
    const int tile = blockIdx.x * 8 + wave;
    if (tile * 16 >= N) return;             // uniform per wave -> EXEC stays all-ones
    const int half = lane >> 4;
    const int m    = lane & 15;
    const int ncol = tile * 16 + (lane & 15);

    const _Float16* A = w16 + (long long)img * 16 * Kpad + (long long)m * Kpad;
    const _Float16* B = bt + (long long)img * N * Kpad + (long long)ncol * Kpad;

    v8f acc;
    #pragma unroll
    for (int r = 0; r < 8; ++r) acc[r] = 0.f;

    for (int k0 = 0; k0 < Kpad; k0 += 32) {
        // A 16x32 f16 layout: lane m, VGPRs 0-3 hold K = half*8 + 0..7, VGPRs 4-7 hold K = 16+half*8 + 0..7
        v8h a0 = *(const v8h*)(A + k0 + half * 8);
        v8h a1 = *(const v8h*)(A + k0 + 16 + half * 8);
        // B 32x16 f16 layout: lane n, VGPRs 0-3 hold K = half*8 + 0..7, VGPRs 4-7 hold K = 16+half*8 + 0..7
        v8h b0 = *(const v8h*)(B + k0 + half * 8);
        v8h b1 = *(const v8h*)(B + k0 + 16 + half * 8);
        v16h a = __builtin_shufflevector(a0, a1, 0,1,2,3,4,5,6,7,8,9,10,11,12,13,14,15);
        v16h b = __builtin_shufflevector(b0, b1, 0,1,2,3,4,5,6,7,8,9,10,11,12,13,14,15);
        acc = __builtin_amdgcn_wmma_f32_16x16x32_f16(false, a, false, b,
                                                     (short)0, acc, false, false);
    }

    const float* Bb = (img < 8) ? (ab + img * 16) : ob;
    float* o = out + (long long)img * 32 * N;
    #pragma unroll
    for (int r = 0; r < 8; ++r) {
        int mm = r + half * 8;              // C/D: VGPR r, lanes 0-15 -> M=r, 16-31 -> M=r+8
        float v = acc[r] + Bb[mm];
        o[(long long)mm * N + ncol]        = v > 0.f ? v  : 0.f;  // relu(x)
        o[(long long)(mm + 16) * N + ncol] = v < 0.f ? -v : 0.f;  // relu(-x)
    }
}

// ---------------- xin = concat(feat, prev_emb, msg_recp) per agent ----------------
__global__ void xin_kernel(const float* __restrict__ feat, const float* __restrict__ prev_emb,
                           const float* __restrict__ mrecp, float* __restrict__ xin) {
    int a = blockIdx.x, t = threadIdx.x;
    float* xo = xin + (long long)a * LSTMIN;
    const float* f = feat + (long long)a * FEAT;
    for (int k = t; k < FEAT; k += blockDim.x) xo[k] = f[k];
    for (int k = t; k < EMB; k += blockDim.x) xo[FEAT + k] = prev_emb[a * EMB + k];
    if (t < AGENTS - 1) xo[FEAT + EMB + t] = mrecp[a * (AGENTS - 1) + t];
}

// ---------------- LSTM gates: gates[j] = Wih[j]·x + Whh[j]·h + bih[j] + bhh[j] ----------------
__global__ __launch_bounds__(256) void gates_kernel(
        const float* __restrict__ xin, const float* __restrict__ feat_omni,
        const float* __restrict__ hx, const float* __restrict__ ohx,
        const float* __restrict__ wih, const float* __restrict__ whh,
        const float* __restrict__ bih, const float* __restrict__ bhh,
        const float* __restrict__ o_wih, const float* __restrict__ o_whh,
        const float* __restrict__ o_bih, const float* __restrict__ o_bhh,
        float* __restrict__ gates) {
    int img  = blockIdx.y;
    int wave = threadIdx.x >> 5, lane = threadIdx.x & 31;
    int j = blockIdx.x * 8 + wave;           // 0..2047
    const float *x, *h, *Wi, *Wh, *bi, *bh;
    int Kx;
    if (img < 8) {
        x  = xin + (long long)img * LSTMIN;  Kx = LSTMIN;
        h  = hx + (long long)img * HID;
        Wi = wih + (long long)img * 4 * HID * LSTMIN + (long long)j * LSTMIN;
        Wh = whh + (long long)img * 4 * HID * HID + (long long)j * HID;
        bi = bih + (long long)img * 4 * HID;
        bh = bhh + (long long)img * 4 * HID;
    } else {
        x  = feat_omni;  Kx = FEAT;  h = ohx;
        Wi = o_wih + (long long)j * FEAT;
        Wh = o_whh + (long long)j * HID;
        bi = o_bih;  bh = o_bhh;
    }
    float acc = 0.f;
    for (int k = lane; k < Kx; k += 32)       // rows have odd stride -> coalesced dword stream
        acc += x[k] * __builtin_nontemporal_load(&Wi[k]);
    {   // Whh rows are 16B aligned -> b128 stream
        const v4f* Wh4 = (const v4f*)Wh;
        const v4f* h4  = (const v4f*)h;
        #pragma unroll
        for (int i = 0; i < 4; ++i) {
            int k = lane + 32 * i;
            v4f wv = __builtin_nontemporal_load(&Wh4[k]);
            v4f hv = h4[k];
            acc += wv.x * hv.x + wv.y * hv.y + wv.z * hv.z + wv.w * hv.w;
        }
    }
    #pragma unroll
    for (int off = 16; off; off >>= 1) acc += __shfl_xor(acc, off, 32);
    if (lane == 0) gates[(long long)img * 4 * HID + j] = acc + bi[j] + bh[j];
}

// ---------------- LSTM pointwise: gate order i, f, g, o ----------------
__global__ __launch_bounds__(512) void lstm_pw(
        const float* __restrict__ gates, const float* __restrict__ cx, const float* __restrict__ ocx,
        float* __restrict__ hs, float* __restrict__ cs,
        float* __restrict__ oh, float* __restrict__ oc) {
    int img = blockIdx.x, u = threadIdx.x;
    const float* g = gates + (long long)img * 4 * HID;
    float gi = g[u], gf = g[HID + u], gg = g[2 * HID + u], go = g[3 * HID + u];
    float cprev = (img < 8) ? cx[img * HID + u] : ocx[u];
    float c2 = sigmoidf_(gf) * cprev + sigmoidf_(gi) * tanhf(gg);
    float h2 = sigmoidf_(go) * tanhf(c2);
    if (img < 8) { hs[img * HID + u] = h2; cs[img * HID + u] = c2; }
    else         { oh[u] = h2;            oc[u] = c2; }
}

// ---------------- small heads: actor, recp (sigmoid), omni critic ----------------
__global__ __launch_bounds__(256) void heads_kernel(
        const float* __restrict__ hs, const float* __restrict__ oh,
        const float* __restrict__ actw, const float* __restrict__ actb,
        const float* __restrict__ recpw,
        const float* __restrict__ ocritw, const float* __restrict__ ocritb,
        float* __restrict__ actors, float* __restrict__ recps, float* __restrict__ ovals) {
    int img = blockIdx.x;
    int wave = threadIdx.x >> 5, lane = threadIdx.x & 31;
    if (img < 8) {
        const float* h = hs + img * HID;
        for (int row = wave; row < NACT + (AGENTS - 1); row += 8) {
            const float* w = (row < NACT)
                ? (actw + (long long)img * NACT * HID + (long long)row * HID)
                : (recpw + (long long)img * (AGENTS - 1) * HID + (long long)(row - NACT) * HID);
            float acc = 0.f;
            for (int k = lane; k < HID; k += 32) acc += h[k] * w[k];
            #pragma unroll
            for (int off = 16; off; off >>= 1) acc += __shfl_xor(acc, off, 32);
            if (lane == 0) {
                if (row < NACT) actors[img * NACT + row] = acc + actb[img * NACT + row];
                else            recps[img * (AGENTS - 1) + (row - NACT)] = sigmoidf_(acc);
            }
        }
    } else {
        for (int row = wave; row < AGENTS; row += 8) {
            const float* w = ocritw + (long long)row * HID;
            float acc = 0.f;
            for (int k = lane; k < HID; k += 32) acc += oh[k] * w[k];
            #pragma unroll
            for (int off = 16; off; off >>= 1) acc += __shfl_xor(acc, off, 32);
            if (lane == 0) ovals[row] = acc + ocritb[row];
        }
    }
}

// ---------------- message logits: logits[a][v] = h2[a]·msgw[a][v] + gumbel[a][v] ----------------
// 524 MB single-use weight stream: b128 nontemporal loads, one wave per output row.
__global__ __launch_bounds__(256) void msglogit_kernel(
        const float* __restrict__ hs, const float* __restrict__ msgw,
        const float* __restrict__ gumbel, float* __restrict__ logits) {
    int a = blockIdx.y;
    int wave = threadIdx.x >> 5, lane = threadIdx.x & 31;
    __shared__ __align__(16) float h[HID];
    for (int k = threadIdx.x; k < HID; k += 256) h[k] = hs[a * HID + k];
    __syncthreads();
    int v = blockIdx.x * 8 + wave;
    const v4f* w4 = (const v4f*)(msgw + (long long)a * VOCAB * HID + (long long)v * HID);
    const v4f* h4 = (const v4f*)h;
    float acc = 0.f;
    #pragma unroll
    for (int i = 0; i < 4; ++i) {
        int k = lane + 32 * i;
        v4f wv = __builtin_nontemporal_load(&w4[k]);
        v4f hv = h4[k];
        acc += wv.x * hv.x + wv.y * hv.y + wv.z * hv.z + wv.w * hv.w;
    }
    #pragma unroll
    for (int off = 16; off; off >>= 1) acc += __shfl_xor(acc, off, 32);
    if (lane == 0) logits[(long long)a * VOCAB + v] = acc + gumbel[(long long)a * VOCAB + v];
}

// ---------------- straight-through gumbel: msgs = one_hot(argmax(logits+g)) exactly -------------
__global__ __launch_bounds__(1024) void argmax_onehot(const float* __restrict__ logits,
                                                      float* __restrict__ msgs) {
    int a = blockIdx.x, t = threadIdx.x;
    const float* L = logits + (long long)a * VOCAB;
    float best = -__builtin_inff();
    int bidx = 0x7fffffff;
    for (int v = t; v < VOCAB; v += 1024) {
        float x = L[v];
        if (x > best) { best = x; bidx = v; }   // v strictly increasing -> first max kept
    }
    __shared__ float bv[1024];
    __shared__ int   bi[1024];
    bv[t] = best; bi[t] = bidx;
    __syncthreads();
    for (int s = 512; s; s >>= 1) {
        if (t < s) {
            if (bv[t + s] > bv[t] || (bv[t + s] == bv[t] && bi[t + s] < bi[t])) {
                bv[t] = bv[t + s]; bi[t] = bi[t + s];
            }
        }
        __syncthreads();
    }
    int am = bi[0];
    for (int v = t; v < VOCAB; v += 1024)
        msgs[(long long)a * VOCAB + v] = (v == am) ? 1.f : 0.f;
}

// =================================================================================
extern "C" void kernel_launch(void* const* d_in, const int* in_sizes, int n_in,
                              void* d_out, int out_size, void* d_ws, size_t ws_size,
                              hipStream_t stream) {
    const float* omni_state = (const float*)d_in[0];
    const float* omni_hx    = (const float*)d_in[1];
    const float* omni_cx    = (const float*)d_in[2];
    const float* obs        = (const float*)d_in[3];
    const float* msg_recp   = (const float*)d_in[4];
    const float* msg_prev   = (const float*)d_in[5];
    /* d_in[6] = msg_recv: dead code in the reference (msg_embeds dropped) */
    const float* hx         = (const float*)d_in[7];
    const float* cx         = (const float*)d_in[8];
    const float* gumbel     = (const float*)d_in[9];
    const float* c1w = (const float*)d_in[10]; const float* c1b = (const float*)d_in[11];
    const float* c2w = (const float*)d_in[12]; const float* c2b = (const float*)d_in[13];
    const float* c3w = (const float*)d_in[14]; const float* c3b = (const float*)d_in[15];
    const float* c4w = (const float*)d_in[16]; const float* c4b = (const float*)d_in[17];
    const float* wih = (const float*)d_in[18]; const float* whh = (const float*)d_in[19];
    const float* bih = (const float*)d_in[20]; const float* bhh = (const float*)d_in[21];
    const float* embw   = (const float*)d_in[22];
    const float* actw   = (const float*)d_in[23]; const float* actb = (const float*)d_in[24];
    const float* recpw  = (const float*)d_in[25];
    const float* msgw   = (const float*)d_in[26];
    const float* oc1w = (const float*)d_in[27]; const float* oc1b = (const float*)d_in[28];
    const float* oc2w = (const float*)d_in[29]; const float* oc2b = (const float*)d_in[30];
    const float* oc3w = (const float*)d_in[31]; const float* oc3b = (const float*)d_in[32];
    const float* oc4w = (const float*)d_in[33]; const float* oc4b = (const float*)d_in[34];
    const float* o_wih = (const float*)d_in[35]; const float* o_whh = (const float*)d_in[36];
    const float* o_bih = (const float*)d_in[37]; const float* o_bhh = (const float*)d_in[38];
    const float* ocritw = (const float*)d_in[39]; const float* ocritb = (const float*)d_in[40];

    float* out = (float*)d_out;
    float* actors = out;                    // 8*6
    float* recps  = out + 48;               // 8*7
    float* msgs   = out + 104;              // 8*32000
    float* hs     = out + 256104;           // 8*512
    float* cs     = out + 260200;           // 8*512
    float* ovals  = out + 264296;           // 8
    float* oh     = out + 264304;           // 512
    float* oc     = out + 264816;           // 512

    // ---- workspace carve-up (256B aligned regions) ----
    char* wsb = (char*)d_ws;
    size_t off = 0;
    auto carve = [&](size_t bytes) -> char* {
        char* p = wsb + off;
        off = (off + bytes + 255) & ~(size_t)255;
        return p;
    };
    float*    buf1  = (float*)carve(9ULL * 32 * 4096 * 4);     // conv1 out (CReLU)
    float*    buf2  = (float*)carve(9ULL * 32 * 1024 * 4);     // conv2 out
    float*    buf3  = (float*)carve(9ULL * 32 * 256 * 4);      // conv3 out
    float*    feat  = (float*)carve(9ULL * FEAT * 4);          // conv4 out = features
    float*    prev  = (float*)carve(8ULL * EMB * 4);
    float*    xin   = (float*)carve(8ULL * LSTMIN * 4);
    float*    gates = (float*)carve(9ULL * 4 * HID * 4);
    float*    logit = (float*)carve(8ULL * VOCAB * 4);
    _Float16* w16   = (_Float16*)carve(9ULL * 16 * 288 * 2);   // f16 weights (reused per layer)
    _Float16* bt    = (_Float16*)carve(9ULL * 1024 * 288 * 2); // transposed im2col (max: layer 2)

    // prev_emb accumulated via atomics -> zero it each call
    zero_kernel<<<8, 256, 0, stream>>>(prev, AGENTS * EMB);
    emb_kernel<<<dim3(8, AGENTS), 256, 0, stream>>>(msg_prev, embw, prev);

    // ---- CNN: 9 images (8 agents + omni), 4 layers of wcvt -> im2col -> WMMA GEMM+CReLU ----
    // L1: Cin=3 (K=27, Kpad=32), 128->64, N=4096
    wcvt_kernel<<<(9 * 16 * 32) / 256, 256, 0, stream>>>(c1w, oc1w, w16, 27, 32);
    im2col_kernel<<<dim3((4096 * 32) / 256, 9), 256, 0, stream>>>(
        obs, 3LL * 128 * 128, omni_state, bt, 3, 128, 128, 64, 27, 32, 4096);
    gemm_crelu_wmma<<<dim3(32, 9), 256, 0, stream>>>(w16, bt, c1b, oc1b, buf1, 4096, 32);
    // L2: Cin=32 (K=288), 64->32, N=1024
    wcvt_kernel<<<(9 * 16 * 288) / 256, 256, 0, stream>>>(c2w, oc2w, w16, 288, 288);
    im2col_kernel<<<dim3((1024 * 288) / 256, 9), 256, 0, stream>>>(
        buf1, 32LL * 64 * 64, buf1 + 8LL * 32 * 64 * 64, bt, 32, 64, 64, 32, 288, 288, 1024);
    gemm_crelu_wmma<<<dim3(8, 9), 256, 0, stream>>>(w16, bt, c2b, oc2b, buf2, 1024, 288);
    // L3: 32->16, N=256
    wcvt_kernel<<<(9 * 16 * 288) / 256, 256, 0, stream>>>(c3w, oc3w, w16, 288, 288);
    im2col_kernel<<<dim3((256 * 288) / 256, 9), 256, 0, stream>>>(
        buf2, 32LL * 32 * 32, buf2 + 8LL * 32 * 32 * 32, bt, 32, 32, 32, 16, 288, 288, 256);
    gemm_crelu_wmma<<<dim3(2, 9), 256, 0, stream>>>(w16, bt, c3b, oc3b, buf3, 256, 288);
    // L4: 16->8, N=64
    wcvt_kernel<<<(9 * 16 * 288) / 256, 256, 0, stream>>>(c4w, oc4w, w16, 288, 288);
    im2col_kernel<<<dim3((64 * 288) / 256, 9), 256, 0, stream>>>(
        buf3, 32LL * 16 * 16, buf3 + 8LL * 32 * 16 * 16, bt, 32, 16, 16, 8, 288, 288, 64);
    gemm_crelu_wmma<<<dim3(1, 9), 256, 0, stream>>>(w16, bt, c4b, oc4b, feat, 64, 288);

    xin_kernel<<<AGENTS, 512, 0, stream>>>(feat, prev, msg_recp, xin);

    gates_kernel<<<dim3(256, 9), 256, 0, stream>>>(
        xin, feat + 8LL * FEAT, hx, omni_hx,
        wih, whh, bih, bhh, o_wih, o_whh, o_bih, o_bhh, gates);

    lstm_pw<<<9, 512, 0, stream>>>(gates, cx, omni_cx, hs, cs, oh, oc);

    heads_kernel<<<9, 256, 0, stream>>>(hs, oh, actw, actb, recpw, ocritw, ocritb,
                                        actors, recps, ovals);

    msglogit_kernel<<<dim3(VOCAB / 8, AGENTS), 256, 0, stream>>>(hs, msgw, gumbel, logit);

    argmax_onehot<<<AGENTS, 1024, 0, stream>>>(logit, msgs);
}
